// AttnDecoderRNN_68642167324794
// MI455X (gfx1250) — matrix operand order repaired and checked by reading, not compiled, not run on hardware
//
#include <hip/hip_runtime.h>
#include <math.h>

typedef __attribute__((ext_vector_type(16))) _Float16 v16h;
typedef __attribute__((ext_vector_type(8)))  _Float16 v8h;
typedef __attribute__((ext_vector_type(8)))  float    v8f;

#define B_  4096
#define S_  48
#define H_  128
#define O_  32
#define L_  48

// ---------------------------------------------------------------------------
// WMMA fragment helpers (v_wmma_f32_16x16x32_f16), layouts per CDNA5 ISA 7.12.2
// All operands are f16 at rest -> fragment loads are pure b128 vector loads.
// ---------------------------------------------------------------------------

// A fragment: 16x32 tile of a row-major f16 matrix (A points at tile origin).
// lanes 0-15: M=lane, elems 0..7 -> K=0..7,  8..15 -> K=16..23
// lanes 16-31: M=lane-16, elems 0..7 -> K=8..15, 8..15 -> K=24..31
static __device__ __forceinline__ v16h load_a16x32_h(const _Float16* A, int lda) {
  const int lane = threadIdx.x & 31;
  const int m  = lane & 15;
  const int kh = (lane >> 4) << 3;   // 0 or 8
  const _Float16* p = A + m * lda + kh;
  v8h lo = *(const v8h*)(p);
  v8h hi = *(const v8h*)(p + 16);
  v16h a;
#pragma unroll
  for (int i = 0; i < 8; ++i) { a[i] = lo[i]; a[i + 8] = hi[i]; }
  return a;
}

// B fragment: 32x16 tile with B[k][n] = W[n][k]; W row-major (N x K) f16,
// W points at (n0, k0). lanes 0-15: N=lane, K=0..7,16..23; lanes 16-31: K=8..15,24..31.
static __device__ __forceinline__ v16h load_bT32x16_h(const _Float16* W, int ldw) {
  const int lane = threadIdx.x & 31;
  const int n  = lane & 15;
  const int kb = (lane >> 4) << 3;   // 0 or 8
  const _Float16* p = W + n * ldw + kb;
  v8h lo = *(const v8h*)(p);
  v8h hi = *(const v8h*)(p + 16);
  v16h b;
#pragma unroll
  for (int i = 0; i < 8; ++i) { b[i] = lo[i]; b[i + 8] = hi[i]; }
  return b;
}

static __device__ __forceinline__ v8f wmma_acc(v16h a, v16h b, v8f c) {
  return __builtin_amdgcn_wmma_f32_16x16x32_f16(false, a, false, b,
                                                (short)0, c, false, false);
}

// One 16x16 f32 tile: C = A(16xK) * W^T(Kx16), all-f16 operands, f32 accum.
static __device__ __forceinline__ v8f gemm_tile_h(const _Float16* A, int lda,
                                                  const _Float16* W, int ldw, int K) {
  v8f acc = {};
  for (int k = 0; k < K; k += 32)
    acc = wmma_acc(load_a16x32_h(A + k, lda), load_bT32x16_h(W + k, ldw), acc);
  return acc;
}

// C/D layout: VGPR r, lanes 0-15: M=r, N=lane; lanes 16-31: M=8+r, N=lane-16
static __device__ __forceinline__ void store_c16x16(float* C, int ldc, v8f c,
                                                    const float* bias /*offset n0*/) {
  const int lane = threadIdx.x & 31;
  const int n  = lane & 15;
  const int m0 = (lane >> 4) << 3;
  const float bb = bias ? bias[n] : 0.0f;
#pragma unroll
  for (int r = 0; r < 8; ++r)
    C[(m0 + r) * ldc + n] = c[r] + bb;
}

static __device__ __forceinline__ void store_c16x16_f16(_Float16* C, int ldc, v8f c,
                                                        const float* bias) {
  const int lane = threadIdx.x & 31;
  const int n  = lane & 15;
  const int m0 = (lane >> 4) << 3;
  const float bb = bias ? bias[n] : 0.0f;
#pragma unroll
  for (int r = 0; r < 8; ++r)
    C[(m0 + r) * ldc + n] = (_Float16)(c[r] + bb);
}

static __device__ __forceinline__ float wave_sum(float v) {
#pragma unroll
  for (int off = 16; off > 0; off >>= 1) v += __shfl_xor(v, off, 32);
  return v;
}
static __device__ __forceinline__ float wave_max(float v) {
#pragma unroll
  for (int off = 16; off > 0; off >>= 1) v = fmaxf(v, __shfl_xor(v, off, 32));
  return v;
}
static __device__ __forceinline__ float sigmoidf_(float x) {
  return 1.0f / (1.0f + __expf(-x));
}

// ---------------------------------------------------------------------------
// Kernels
// ---------------------------------------------------------------------------

// generic f32 -> f16 convert
__global__ void dec_cvt_k(const float* __restrict__ src, _Float16* __restrict__ dst,
                          int n) {
  int i = blockIdx.x * 256 + threadIdx.x;
  if (i < n) dst[i] = (_Float16)src[i];
}

// h = enc_hidden[0] (f32 + f16 copies); act = SOS(0); dur = 0
__global__ void dec_init_k(const float* __restrict__ eh, float* __restrict__ h,
                           _Float16* __restrict__ h16, int* __restrict__ act,
                           float* __restrict__ dur) {
  int i = blockIdx.x * 256 + threadIdx.x;
  if (i < B_ * H_) { float v = eh[i]; h[i] = v; h16[i] = (_Float16)v; }
  if (i < B_) { act[i] = 0; dur[i] = 0.0f; }
}

// k_proj[b,s,g] = sum_h enc[b,s,h]*Ua[g,h] + bu[g]   (f16 in, f16 out)
__global__ void dec_kproj_k(const _Float16* __restrict__ enc16,
                            const _Float16* __restrict__ Ua16,
                            const float* __restrict__ bu, _Float16* __restrict__ kp) {
  const int w  = threadIdx.x >> 5;
  const int n0 = w * 16;
  const int m0 = blockIdx.x * 16;
  v8f acc = gemm_tile_h(enc16 + (size_t)m0 * H_, H_, Ua16 + n0 * H_, H_, H_);
  store_c16x16_f16(kp + (size_t)m0 * H_ + n0, H_, acc, bu + n0);
}

// q = h @ Wa^T + ba
__global__ void dec_q_k(const _Float16* __restrict__ h16,
                        const _Float16* __restrict__ Wa16,
                        const float* __restrict__ ba, float* __restrict__ q) {
  const int w  = threadIdx.x >> 5;
  const int n0 = w * 16;
  const int m0 = blockIdx.x * 16;
  v8f acc = gemm_tile_h(h16 + (size_t)m0 * H_, H_, Wa16 + n0 * H_, H_, H_);
  store_c16x16(q + (size_t)m0 * H_ + n0, H_, acc, ba + n0);
}

// Attention: scores -> softmax -> ctx; builds x16 = [emb[act], dur, ctx];
// writes attention weights to the output. One wave per batch row, 8 rows/block.
__global__ void dec_attn_k(const float* __restrict__ q, const _Float16* __restrict__ kp,
                           const _Float16* __restrict__ enc16,
                           const float* __restrict__ emb,
                           const float* __restrict__ Va, const float* __restrict__ bv,
                           const int* __restrict__ act, const float* __restrict__ dur,
                           _Float16* __restrict__ x16, float* __restrict__ attn_out,
                           int l) {
  __shared__ float sc[8][S_];
  const int w    = threadIdx.x >> 5;
  const int lane = threadIdx.x & 31;
  const int b    = blockIdx.x * 8 + w;

  float qv[4], va[4];
#pragma unroll
  for (int j = 0; j < 4; ++j) {
    qv[j] = q[(size_t)b * H_ + lane + j * 32];
    va[j] = Va[lane + j * 32];
  }

  for (int s = 0; s < S_; ++s) {
    const _Float16* kpr = kp + ((size_t)b * S_ + s) * H_;
    float acc = 0.0f;
#pragma unroll
    for (int j = 0; j < 4; ++j)
      acc += va[j] * tanhf(qv[j] + (float)kpr[lane + j * 32]);
    acc = wave_sum(acc);
    if (lane == 0) sc[w][s] = acc + bv[0];
  }
  __syncthreads();

  // softmax over S=48: lane covers s=lane and (lane<16) s=lane+32
  float v0 = sc[w][lane];
  float v1 = (lane < 16) ? sc[w][32 + lane] : -3.4e38f;
  float mx = wave_max(fmaxf(v0, v1));
  float e0 = __expf(v0 - mx);
  float e1 = (lane < 16) ? __expf(v1 - mx) : 0.0f;
  float sm = wave_sum(e0 + e1);
  float w0 = e0 / sm, w1 = e1 / sm;
  sc[w][lane] = w0;
  if (lane < 16) sc[w][32 + lane] = w1;
  attn_out[((size_t)b * L_ + l) * S_ + lane] = w0;
  if (lane < 16) attn_out[((size_t)b * L_ + l) * S_ + 32 + lane] = w1;
  __syncthreads();

  // ctx[h] = sum_s w[s] * enc[b,s,h]   (enc read from the f16, L2-resident copy)
  float ctx[4] = {0.f, 0.f, 0.f, 0.f};
  for (int s = 0; s < S_; ++s) {
    float ws = sc[w][s];
    const _Float16* er = enc16 + ((size_t)b * S_ + s) * H_;
#pragma unroll
    for (int j = 0; j < 4; ++j) ctx[j] += ws * (float)er[lane + j * 32];
  }

  // x = [emb[act[b]] (127), dur (1), ctx (128)], stored f16 for the W_ih GEMM
  _Float16* xb = x16 + (size_t)b * 256;
  const float* er = emb + (size_t)act[b] * (H_ - 1);
#pragma unroll
  for (int j = 0; j < 4; ++j) {
    int idx = lane + j * 32;
    xb[idx] = (_Float16)((idx == 127) ? dur[b] : er[idx]);
    xb[128 + idx] = (_Float16)ctx[j];
  }
}

// GRU cell + output head. Block = 256 thr (8 waves) owns 16 batch rows.
__global__ void dec_gru_k(const _Float16* __restrict__ x16, float* __restrict__ h,
                          _Float16* __restrict__ h16,
                          const _Float16* __restrict__ Wih16,
                          const float* __restrict__ b_ih,
                          const _Float16* __restrict__ Whh16,
                          const float* __restrict__ b_hh,
                          const _Float16* __restrict__ Wo16,
                          const float* __restrict__ bo,
                          float* __restrict__ logits_raw, int* __restrict__ act,
                          float* __restrict__ dur, int l) {
  __shared__ float gi[16][3 * H_];
  __shared__ float gh[16][3 * H_];
  __shared__ __align__(16) _Float16 h2s[16][H_];
  __shared__ float ls[16][O_];

  const int w  = threadIdx.x >> 5;
  const int bm = blockIdx.x * 16;

  // Hoist A fragments once; reuse across this wave's 3 N-tiles.
  v16h ax[8];
#pragma unroll
  for (int kt = 0; kt < 8; ++kt)
    ax[kt] = load_a16x32_h(x16 + (size_t)bm * 256 + kt * 32, 256);
  v16h ah[4];
#pragma unroll
  for (int kt = 0; kt < 4; ++kt)
    ah[kt] = load_a16x32_h(h16 + (size_t)bm * H_ + kt * 32, H_);

  // gi = x @ W_ih^T + b_ih ; gh = h @ W_hh^T + b_hh   (N=384 -> 24 tiles, 3/wave)
#pragma unroll
  for (int t = 0; t < 3; ++t) {
    const int n0 = (w * 3 + t) * 16;
    v8f acc = {};
#pragma unroll
    for (int kt = 0; kt < 8; ++kt)
      acc = wmma_acc(ax[kt], load_bT32x16_h(Wih16 + n0 * 256 + kt * 32, 256), acc);
    store_c16x16(&gi[0][0] + n0, 3 * H_, acc, b_ih + n0);
    v8f acc2 = {};
#pragma unroll
    for (int kt = 0; kt < 4; ++kt)
      acc2 = wmma_acc(ah[kt], load_bT32x16_h(Whh16 + n0 * H_ + kt * 32, H_), acc2);
    store_c16x16(&gh[0][0] + n0, 3 * H_, acc2, b_hh + n0);
  }
  __syncthreads();

  // gates
  for (int i = threadIdx.x; i < 16 * H_; i += 256) {
    int r = i >> 7, c = i & (H_ - 1);
    float rr = sigmoidf_(gi[r][c] + gh[r][c]);
    float zz = sigmoidf_(gi[r][H_ + c] + gh[r][H_ + c]);
    float nn = tanhf(gi[r][2 * H_ + c] + rr * gh[r][2 * H_ + c]);
    float hold = h[(size_t)(bm + r) * H_ + c];
    float hnew = (1.0f - zz) * nn + zz * hold;
    h2s[r][c] = (_Float16)hnew;
    h[(size_t)(bm + r) * H_ + c] = hnew;
    h16[(size_t)(bm + r) * H_ + c] = (_Float16)hnew;
  }
  __syncthreads();

  // logits = h2 @ Wo^T + bo  (O=32 -> 2 tiles on waves 0,1; A from LDS f16)
  if (w < 2) {
    const int n0 = w * 16;
    v8f acc = gemm_tile_h(&h2s[0][0], H_, Wo16 + n0 * H_, H_, H_);
    store_c16x16(&ls[0][0] + n0, O_, acc, bo + n0);
  }
  __syncthreads();

  for (int i = threadIdx.x; i < 16 * O_; i += 256) {
    int r = i >> 5, c = i & (O_ - 1);
    logits_raw[((size_t)(bm + r) * L_ + l) * O_ + c] = ls[r][c];
  }
  if (threadIdx.x < 16) {
    int r = threadIdx.x;
    float best = ls[r][0];
    int bi = 0;
    for (int c = 1; c < O_ - 1; ++c) {
      float v = ls[r][c];
      if (v > best) { best = v; bi = c; }
    }
    act[bm + r] = bi;
    dur[bm + r] = sigmoidf_(ls[r][O_ - 1]);
  }
}

// log_softmax / softmax over first 31 logits + sigmoid duration.
// Raw logits were staged in the log_prob output region; transformed in place.
__global__ void dec_final_k(float* __restrict__ logp, float* __restrict__ prob) {
  const int w    = threadIdx.x >> 5;
  const int lane = threadIdx.x & 31;
  const size_t row = (size_t)blockIdx.x * 8 + w;   // row = b*L + l
  float* raw = logp + row * O_;
  float rv = raw[lane];
  float v  = (lane < O_ - 1) ? rv : -3.4e38f;
  float mx = wave_max(v);
  float e  = (lane < O_ - 1) ? __expf(v - mx) : 0.0f;
  float sm = wave_sum(e);
  if (lane < O_ - 1) {
    prob[row * O_ + lane] = e / sm;
    raw[lane] = v - mx - __logf(sm);
  } else {
    float d = sigmoidf_(rv);
    prob[row * O_ + lane] = d;
    raw[lane] = d;
  }
}

__global__ void dec_copyh_k(const float* __restrict__ h, float* __restrict__ hf) {
  int i = blockIdx.x * 256 + threadIdx.x;
  if (i < B_ * H_) hf[i] = h[i];
}

// ---------------------------------------------------------------------------
// Launch
// ---------------------------------------------------------------------------
extern "C" void kernel_launch(void* const* d_in, const int* in_sizes, int n_in,
                              void* d_out, int out_size, void* d_ws, size_t ws_size,
                              hipStream_t stream) {
  (void)in_sizes; (void)n_in; (void)out_size; (void)ws_size;

  const float* enc   = (const float*)d_in[1];   // (B,S,H)
  const float* eh    = (const float*)d_in[2];   // (1,B,H)
  const float* emb   = (const float*)d_in[4];   // (VOCAB,H-1)
  const float* Wa    = (const float*)d_in[5];
  const float* ba    = (const float*)d_in[6];
  const float* Ua    = (const float*)d_in[7];
  const float* bu    = (const float*)d_in[8];
  const float* Va    = (const float*)d_in[9];
  const float* bv    = (const float*)d_in[10];
  const float* W_ih  = (const float*)d_in[11];  // (384,256)
  const float* W_hh  = (const float*)d_in[12];  // (384,128)
  const float* b_ih  = (const float*)d_in[13];
  const float* b_hh  = (const float*)d_in[14];
  const float* Wo    = (const float*)d_in[15];  // (32,128)
  const float* bo    = (const float*)d_in[16];

  float* out  = (float*)d_out;
  float* logp = out;                                    // B*L*O
  float* prob = out + (size_t)B_ * L_ * O_;             // B*L*O
  float* hf   = prob + (size_t)B_ * L_ * O_;            // B*H
  float* attn = hf + (size_t)B_ * H_;                   // B*L*S

  // workspace layout (f16 tensors first; everything 16B aligned)
  _Float16* kp    = (_Float16*)d_ws;                    // B*S*H
  _Float16* enc16 = kp + (size_t)B_ * S_ * H_;          // B*S*H
  _Float16* ua16  = enc16 + (size_t)B_ * S_ * H_;       // H*H
  _Float16* wa16  = ua16 + H_ * H_;                     // H*H
  _Float16* wih16 = wa16 + H_ * H_;                     // 384*256
  _Float16* whh16 = wih16 + 3 * H_ * 2 * H_;            // 384*128
  _Float16* wo16  = whh16 + 3 * H_ * H_;                // 32*128
  _Float16* h16   = wo16 + O_ * H_;                     // B*H
  _Float16* x16   = h16 + (size_t)B_ * H_;              // B*256
  float* hst      = (float*)(x16 + (size_t)B_ * 256);   // B*H
  float* qb       = hst + (size_t)B_ * H_;              // B*H
  float* durb     = qb + (size_t)B_ * H_;               // B
  int*   actb     = (int*)(durb + B_);                  // B

  // one-time conversions (weights + encoder outputs) to f16
  dec_cvt_k<<<(B_ * S_ * H_ + 255) / 256, 256, 0, stream>>>(enc, enc16, B_ * S_ * H_);
  dec_cvt_k<<<(H_ * H_ + 255) / 256, 256, 0, stream>>>(Ua, ua16, H_ * H_);
  dec_cvt_k<<<(H_ * H_ + 255) / 256, 256, 0, stream>>>(Wa, wa16, H_ * H_);
  dec_cvt_k<<<(3 * H_ * 2 * H_ + 255) / 256, 256, 0, stream>>>(W_ih, wih16,
                                                               3 * H_ * 2 * H_);
  dec_cvt_k<<<(3 * H_ * H_ + 255) / 256, 256, 0, stream>>>(W_hh, whh16, 3 * H_ * H_);
  dec_cvt_k<<<(O_ * H_ + 255) / 256, 256, 0, stream>>>(Wo, wo16, O_ * H_);

  dec_init_k<<<(B_ * H_ + 255) / 256, 256, 0, stream>>>(eh, hst, h16, actb, durb);
  dec_kproj_k<<<(B_ * S_) / 16, 256, 0, stream>>>(enc16, ua16, bu, kp);

  for (int l = 0; l < L_; ++l) {
    dec_q_k<<<B_ / 16, 256, 0, stream>>>(h16, wa16, ba, qb);
    dec_attn_k<<<B_ / 8, 256, 0, stream>>>(qb, kp, enc16, emb, Va, bv, actb, durb,
                                           x16, attn, l);
    dec_gru_k<<<B_ / 16, 256, 0, stream>>>(x16, hst, h16, wih16, b_ih, whh16, b_hh,
                                           wo16, bo, logp, actb, durb, l);
  }

  dec_final_k<<<(B_ * L_) / 8, 256, 0, stream>>>(logp, prob);
  dec_copyh_k<<<(B_ * H_ + 255) / 256, 256, 0, stream>>>(hst, hf);
}
